// FCGRU_20590073217366
// MI455X (gfx1250) — matrix operand order repaired and checked by reading, not compiled, not run on hardware
//
#include <hip/hip_runtime.h>
#include <stdint.h>

// ---------------- types ----------------
typedef __attribute__((ext_vector_type(16))) __bf16 v16bf;
typedef __attribute__((ext_vector_type(8)))  __bf16 v8bf;
typedef __attribute__((ext_vector_type(8)))  float  v8f;

union ABfrag { v16bf v; v8bf h[2]; };

#define BATCH 256
#define TSTEPS 256
#define IN_DIM 64
#define HID 1024
#define DIM2 512
#define G3 3072   // 3*HID

// ---------------- CDNA5 async-copy helpers (ISA 08_async_tensor) ----------------
// Per-lane 16B copy global -> LDS, tracked with ASYNCcnt (bypasses VGPRs).
__device__ __forceinline__ void async_ld_b128(const void* lds_ptr, const void* gptr) {
    unsigned lds = (unsigned)(uintptr_t)lds_ptr;  // low 32 bits of generic ptr = DS offset
    asm volatile("global_load_async_to_lds_b128 %0, %1, off"
                 :: "v"(lds), "v"((unsigned long long)(uintptr_t)gptr)
                 : "memory");
}
__device__ __forceinline__ void wait_async0() {
    asm volatile("s_wait_asynccnt 0x0" ::: "memory");
}

// ---------------- tiny precompute kernels ----------------
// b12[n] = fc2[n,:] . fc1_b + fc2_b[n]           (n < 1024)
__global__ void bias12_k(const float* __restrict__ fc2w, const float* __restrict__ b1,
                         const float* __restrict__ b2, float* __restrict__ b12) {
    int n = blockIdx.x * blockDim.x + threadIdx.x;
    if (n < HID) {
        float s = 0.f;
        for (int i = 0; i < DIM2; ++i) s += fc2w[n * DIM2 + i] * b1[i];
        b12[n] = s + b2[n];
    }
}
// bg[m] = w_ih[m,:] . b12 + b_ih[m]              (m < 3072)
__global__ void biasg_k(const float* __restrict__ wih, const float* __restrict__ b12,
                        const float* __restrict__ bih, float* __restrict__ bg) {
    int m = blockIdx.x * blockDim.x + threadIdx.x;
    if (m < G3) {
        float s = 0.f;
        for (int i = 0; i < HID; ++i) s += wih[m * HID + i] * b12[i];
        bg[m] = s + bih[m];
    }
}
__global__ void cvt_bf16_k(const float* __restrict__ src, __bf16* __restrict__ dst, int n) {
    int i = blockIdx.x * blockDim.x + threadIdx.x;
    if (i < n) dst[i] = (__bf16)src[i];
}

// ---------------- generic bf16 WMMA GEMM ----------------
// C[M,N] = (reluA? relu(A) : A)[M,K] @ B^T + bias, where B element (n,k) is
// Bm[n*ldbn + k*ldbk]. f32 in/out, bf16 WMMA compute, f32 accumulate.
// Tile: 32(M) x 64(N), BK=64. 256 threads = 8 waves, each wave one 16x16 C tile.
// Requires M%32==0, N%64==0, K%64==0 (true for all uses here).
__global__ __launch_bounds__(256)
void gemm_bf16_k(const float* __restrict__ A, int lda, int reluA,
                 const float* __restrict__ Bm, int ldbn, int ldbk,
                 const float* __restrict__ bias, int reluOut,
                 float* __restrict__ C, int ldc, int K)
{
    __shared__ __align__(16) __bf16 sA[32][80]; // 160B row stride: 16B-aligned, bank-spread
    __shared__ __align__(16) __bf16 sB[64][80]; // N-major: sB[n][k]

    const int tid  = threadIdx.x;
    const int mb   = blockIdx.x * 32;
    const int nb   = blockIdx.y * 64;
    const int lane = tid & 31;
    const int wv   = tid >> 5;
    const int wrow = wv >> 2, wcol = wv & 3;       // 2x4 waves -> 32x64 tile
    const int half = lane >> 4, l16 = lane & 15;

    v8f acc = {0.f, 0.f, 0.f, 0.f, 0.f, 0.f, 0.f, 0.f};

    const int ar = tid >> 3;           // 0..31 (A row)
    const int ak = (tid & 7) * 8;      // A k-offset
    const int bn = tid >> 2;           // 0..63 (B n)
    const int bk = (tid & 3) * 16;     // B k-offset

    for (int k0 = 0; k0 < K; k0 += 64) {
        // ---- stage A tile (f32 -> bf16), 8 elems/thread ----
        const float4* ap = (const float4*)(A + (long)(mb + ar) * lda + k0 + ak);
        float4 va = ap[0], vb = ap[1];
        if (reluA) {
            va.x = fmaxf(va.x, 0.f); va.y = fmaxf(va.y, 0.f);
            va.z = fmaxf(va.z, 0.f); va.w = fmaxf(va.w, 0.f);
            vb.x = fmaxf(vb.x, 0.f); vb.y = fmaxf(vb.y, 0.f);
            vb.z = fmaxf(vb.z, 0.f); vb.w = fmaxf(vb.w, 0.f);
        }
        __bf16* da = &sA[ar][ak];
        da[0] = (__bf16)va.x; da[1] = (__bf16)va.y; da[2] = (__bf16)va.z; da[3] = (__bf16)va.w;
        da[4] = (__bf16)vb.x; da[5] = (__bf16)vb.y; da[6] = (__bf16)vb.z; da[7] = (__bf16)vb.w;
        // ---- stage B tile (strided gather, f32 -> bf16), 16 elems/thread ----
        const float* bp = Bm + (long)(nb + bn) * ldbn + (long)(k0 + bk) * ldbk;
        #pragma unroll
        for (int j = 0; j < 16; ++j) sB[bn][bk + j] = (__bf16)bp[(long)j * ldbk];
        __syncthreads();

        #pragma unroll
        for (int kk = 0; kk < 64; kk += 32) {
            ABfrag a, b;
            a.h[0] = *(const v8bf*)&sA[wrow * 16 + l16][kk + 16 * half];
            a.h[1] = *(const v8bf*)&sA[wrow * 16 + l16][kk + 16 * half + 8];
            b.h[0] = *(const v8bf*)&sB[wcol * 16 + l16][kk + 16 * half];
            b.h[1] = *(const v8bf*)&sB[wcol * 16 + l16][kk + 16 * half + 8];
            acc = __builtin_amdgcn_wmma_f32_16x16x32_bf16(false, a.v, false, b.v,
                                                          (short)0, acc, false, false);
        }
        __syncthreads();
    }

    const int col = nb + wcol * 16 + l16;
    const float bv = bias ? bias[col] : 0.f;
    #pragma unroll
    for (int i = 0; i < 8; ++i) {
        int row = mb + wrow * 16 + i + 8 * half;
        float v = acc[i] + bv;
        if (reluOut) v = fmaxf(v, 0.f);
        C[(long)row * ldc + col] = v;
    }
}

// ---------------- fused GRU step (double-buffered, async B staging) ----------------
// One time step: gh = h @ w_hh^T (K=1024, bf16 weights), gi = x_t @ Wg^T (K=64),
// then gate math + hidden update. Each WG: 32 batch rows x 64 hidden cols x 3 gates.
// B tiles (pure bf16 copies) stream global->LDS via GLOBAL_LOAD_ASYNC_TO_LDS_B128,
// overlapped with WMMA compute on the other LDS buffer.
__global__ __launch_bounds__(256)
void gru_step_k(const float* __restrict__ h_in, float* __restrict__ h_out,
                const float* __restrict__ x, int t,
                const __bf16* __restrict__ whh,  // bf16 (3072,1024)
                const __bf16* __restrict__ wg,   // bf16 (3072,64)
                const float* __restrict__ bhh,   // f32 (3072)
                const float* __restrict__ bg)    // f32 (3072) = w_ih@b12 + b_ih
{
    __shared__ __align__(16) __bf16 sA[2][32][80];
    __shared__ __align__(16) __bf16 sB[2][3][64][80];

    const int tid  = threadIdx.x;
    const int rb   = blockIdx.x * 32;   // batch-row base
    const int jb   = blockIdx.y * 64;   // hidden-col base
    const int lane = tid & 31;
    const int wv   = tid >> 5;
    const int wrow = wv >> 2, wcol = wv & 3;
    const int half = lane >> 4, l16 = lane & 15;

    v8f ah[3], ai[3];
    #pragma unroll
    for (int g = 0; g < 3; ++g) {
        ah[g] = (v8f){0.f,0.f,0.f,0.f,0.f,0.f,0.f,0.f};
        ai[g] = (v8f){0.f,0.f,0.f,0.f,0.f,0.f,0.f,0.f};
    }

    const int ar = tid >> 3, ak = (tid & 7) * 8;

    // issue 6 async b128s/thread: one 3x64x64-bf16 B chunk (24KB) into sB[buf]
    auto stageB = [&](const __bf16* base, int rstride, int k0, int buf) {
        #pragma unroll
        for (int q = 0; q < 6; ++q) {
            const int s  = q * 256 + tid;       // 0..1535 slots of 8 bf16
            const int g  = s >> 9;              // gate
            const int r  = (s >> 3) & 63;       // row within tile
            const int k8 = (s & 7) * 8;         // k offset within chunk
            async_ld_b128(&sB[buf][g][r][k8],
                          base + (long)((g << 10) + jb + r) * rstride + k0 + k8);
        }
    };
    auto cvtStoreA = [&](float4 va, float4 vb, int buf) {
        __bf16* d = &sA[buf][ar][ak];
        d[0] = (__bf16)va.x; d[1] = (__bf16)va.y; d[2] = (__bf16)va.z; d[3] = (__bf16)va.w;
        d[4] = (__bf16)vb.x; d[5] = (__bf16)vb.y; d[6] = (__bf16)vb.z; d[7] = (__bf16)vb.w;
    };
    auto compute = [&](int buf, v8f* acc) {
        #pragma unroll
        for (int kk = 0; kk < 64; kk += 32) {
            ABfrag a;
            a.h[0] = *(const v8bf*)&sA[buf][wrow * 16 + l16][kk + 16 * half];
            a.h[1] = *(const v8bf*)&sA[buf][wrow * 16 + l16][kk + 16 * half + 8];
            #pragma unroll
            for (int g = 0; g < 3; ++g) {
                ABfrag b;
                b.h[0] = *(const v8bf*)&sB[buf][g][wcol * 16 + l16][kk + 16 * half];
                b.h[1] = *(const v8bf*)&sB[buf][g][wcol * 16 + l16][kk + 16 * half + 8];
                acc[g] = __builtin_amdgcn_wmma_f32_16x16x32_bf16(false, a.v, false, b.v,
                                                                 (short)0, acc[g], false, false);
            }
        }
    };

    // ---- phase 1: recurrent GEMM, K = 1024, 16 chunks, double-buffered ----
    {   // prologue: chunk 0 -> buf 0
        const float4* ap = (const float4*)(h_in + (long)(rb + ar) * HID + ak);
        float4 va = ap[0], vb = ap[1];
        stageB(whh, HID, 0, 0);
        cvtStoreA(va, vb, 0);
        wait_async0();
        __syncthreads();
    }
    for (int c = 0; c < 16; ++c) {
        const int buf = c & 1;
        const int nk  = (c + 1) * 64;
        const bool pre = (c + 1) < 16;
        float4 va, vb;
        if (pre) {  // issue next chunk's loads before computing current chunk
            const float4* ap = (const float4*)(h_in + (long)(rb + ar) * HID + nk + ak);
            va = ap[0]; vb = ap[1];
            stageB(whh, HID, nk, buf ^ 1);
        }
        compute(buf, ah);
        if (pre) cvtStoreA(va, vb, buf ^ 1);
        wait_async0();       // own async writes into buf^1 done
        __syncthreads();     // everyone's staging visible -> safe to swap
    }

    // ---- phase 2: fused input projection, K = 64 (folded fc1*fc2*w_ih) ----
    {
        const float4* ap =
            (const float4*)(x + ((long)(rb + ar) * TSTEPS + t) * IN_DIM + ak);
        float4 va = ap[0], vb = ap[1];
        stageB(wg, IN_DIM, 0, 0);
        cvtStoreA(va, vb, 0);
        wait_async0();
        __syncthreads();
        compute(0, ai);
    }

    // ---- gate math + hidden update (C layout: vgpr i -> M = i + 8*half, lane -> N) ----
    const int col = jb + wcol * 16 + l16;
    const float bh_r = bhh[col],          bg_r = bg[col];
    const float bh_z = bhh[HID + col],    bg_z = bg[HID + col];
    const float bh_n = bhh[2*HID + col],  bg_n = bg[2*HID + col];
    #pragma unroll
    for (int i = 0; i < 8; ++i) {
        const int row = rb + wrow * 16 + i + 8 * half;
        const float r = 1.f / (1.f + __expf(-(ah[0][i] + bh_r + ai[0][i] + bg_r)));
        const float z = 1.f / (1.f + __expf(-(ah[1][i] + bh_z + ai[1][i] + bg_z)));
        const float n = tanhf(ai[2][i] + bg_n + r * (ah[2][i] + bh_n));
        const float hold = h_in[(long)row * HID + col];
        h_out[(long)row * HID + col] = (1.f - z) * n + z * hold;
    }
}

// ---------------- per-sample head selection ----------------
__global__ __launch_bounds__(256)
void heads_k(const float* __restrict__ out2, const int* __restrict__ cult,
             const float* __restrict__ hw, const float* __restrict__ hb,
             float* __restrict__ params)
{
    __shared__ float so[DIM2];
    __shared__ float red[256];
    const int b = blockIdx.x, tid = threadIdx.x;
    so[tid]       = out2[(long)b * DIM2 + tid];
    so[tid + 256] = out2[(long)b * DIM2 + 256 + tid];
    __syncthreads();
    const int c = cult[b];
    const int o = tid >> 4, p = tid & 15;
    const float* w = hw + ((long)c * 16 + o) * DIM2;
    float s = 0.f;
    for (int d = p; d < DIM2; d += 16) s += so[d] * w[d];
    red[tid] = s;
    __syncthreads();
    if (p == 0) {
        float tsum = 0.f;
        #pragma unroll
        for (int q = 0; q < 16; ++q) tsum += red[(o << 4) + q];
        params[b * 16 + o] = tsum + hb[c * 16 + o];
    }
}

// ---------------- host launcher ----------------
extern "C" void kernel_launch(void* const* d_in, const int* in_sizes, int n_in,
                              void* d_out, int out_size, void* d_ws, size_t ws_size,
                              hipStream_t stream) {
    (void)in_sizes; (void)n_in; (void)out_size; (void)ws_size;
    const float* input = (const float*)d_in[0];
    const float* hn    = (const float*)d_in[1];
    const int*   cult  = (const int*)d_in[2];
    const float* fc1_w = (const float*)d_in[3];
    const float* fc1_b = (const float*)d_in[4];
    const float* fc2_w = (const float*)d_in[5];
    const float* fc2_b = (const float*)d_in[6];
    const float* w_ih  = (const float*)d_in[7];
    const float* w_hh  = (const float*)d_in[8];
    const float* b_ih  = (const float*)d_in[9];
    const float* b_hh  = (const float*)d_in[10];
    const float* fc3_w = (const float*)d_in[11];
    const float* fc3_b = (const float*)d_in[12];
    const float* hw    = (const float*)d_in[13];
    const float* hb    = (const float*)d_in[14];

    float* params = (float*)d_out;                 // (256,16)
    float* hn_out = (float*)d_out + BATCH * 16;    // (1,256,1024)

    char* ws = (char*)d_ws;
    size_t off = 0;
    auto carve = [&](size_t bytes) -> void* {
        void* p = (void*)(ws + off);
        off += (bytes + 255) & ~(size_t)255;
        return p;
    };
    float*  T1   = (float*)carve((size_t)G3 * DIM2 * 4);     // w_ih @ fc2
    float*  Wgf  = (float*)carve((size_t)G3 * IN_DIM * 4);   // T1 @ fc1
    __bf16* WgB  = (__bf16*)carve((size_t)G3 * IN_DIM * 2);
    __bf16* WhhB = (__bf16*)carve((size_t)G3 * HID * 2);
    float*  b12  = (float*)carve((size_t)HID * 4);
    float*  bg   = (float*)carve((size_t)G3 * 4);
    float*  h0   = (float*)carve((size_t)BATCH * HID * 4);
    float*  h1   = (float*)carve((size_t)BATCH * HID * 4);
    float*  out2 = (float*)carve((size_t)BATCH * DIM2 * 4);

    // fold biases: b12 = fc2@fc1_b + fc2_b ; bg = w_ih@b12 + b_ih
    bias12_k<<<dim3(HID / 256), 256, 0, stream>>>(fc2_w, fc1_b, fc2_b, b12);
    biasg_k<<<dim3(G3 / 256), 256, 0, stream>>>(w_ih, b12, b_ih, bg);

    // fold weights: T1 = w_ih @ fc2 (B(n,k)=fc2[k*512+n]); Wg = T1 @ fc1 (B(n,k)=fc1[k*64+n])
    gemm_bf16_k<<<dim3(G3 / 32, DIM2 / 64), 256, 0, stream>>>(
        w_ih, HID, 0, fc2_w, 1, DIM2, nullptr, 0, T1, DIM2, HID);
    gemm_bf16_k<<<dim3(G3 / 32, IN_DIM / 64), 256, 0, stream>>>(
        T1, DIM2, 0, fc1_w, 1, IN_DIM, nullptr, 0, Wgf, IN_DIM, DIM2);

    // one-time bf16 conversion of the recurrent-loop weight matrices
    cvt_bf16_k<<<dim3((G3 * IN_DIM + 255) / 256), 256, 0, stream>>>(Wgf, WgB, G3 * IN_DIM);
    cvt_bf16_k<<<dim3((G3 * HID + 255) / 256), 256, 0, stream>>>(w_hh, WhhB, G3 * HID);

    // h0 <- hn (don't mutate inputs)
    hipMemcpyAsync(h0, hn, (size_t)BATCH * HID * 4, hipMemcpyDeviceToDevice, stream);

    // 256 dependent steps, ping-pong hidden state
    for (int t = 0; t < TSTEPS; ++t) {
        const float* hi = (t & 1) ? h1 : h0;
        float*       ho = (t & 1) ? h0 : h1;
        gru_step_k<<<dim3(BATCH / 32, HID / 64), 256, 0, stream>>>(
            hi, ho, input, t, WhhB, WgB, b_hh, bg);
    }
    // after 256 steps final h is in h0
    hipMemcpyAsync(hn_out, h0, (size_t)BATCH * HID * 4, hipMemcpyDeviceToDevice, stream);

    // out2 = relu(relu(h) @ fc3^T + fc3_b)
    gemm_bf16_k<<<dim3(BATCH / 32, DIM2 / 64), 256, 0, stream>>>(
        h0, HID, 1, fc3_w, HID, 1, fc3_b, 1, out2, DIM2, HID);

    // params = einsum('bd,bod->bo', out2, heads_w[cidx]) + heads_b[cidx]
    heads_k<<<dim3(BATCH), 256, 0, stream>>>(out2, cult, hw, hb, params);
}
